// MoE_71554155151555
// MI455X (gfx1250) — matrix operand order repaired and checked by reading, not compiled, not run on hardware
//
#include <hip/hip_runtime.h>
#include <hip/hip_bf16.h>

// MoE dense-all-experts forward for MI455X (gfx1250, wave32, WMMA).
//
// out[n,o] = sum_e gate[n,e] * ( relu(x@W1[e]+b1[e]) @ W2[e] + b2[e] )[n,o]
// Rewritten as:  hg_e = gate[:,e] * relu(x@W1[e]+b1[e])   (f16)
//                out  = sum_e gate[:,e]*b2[e]  +  sum_e hg_e @ W2[e]
// All matmuls via v_wmma_f32_16x16x32_f16 (f16 in / f32 accumulate),
// double-buffered LDS, software-pipelined global->LDS staging.

typedef __attribute__((ext_vector_type(16))) _Float16 v16h_t;
typedef __attribute__((ext_vector_type(8)))  _Float16 v8h_t;
typedef __attribute__((ext_vector_type(8)))  float    v8f_t;
typedef __attribute__((ext_vector_type(4)))  float    v4f_t;

#define N_TOK 4096
#define D_IN  1024
#define D_H   4096
#define D_OUT 1024
#define E_NUM 8

#define KC   64    // K-chunk staged per LDS step (2 wmma k-substeps)
#define LDK  72    // padded LDS k-stride in halfs (144B: 16B-aligned, conflict-free frags)

// ---------- fragment loaders (layouts per CDNA5 ISA 7.12.2) ----------
// A 16x32 f16: lane g=lane>>4 holds K in {8g..8g+7} U {16+8g..16+8g+7}, row = lane&15
__device__ __forceinline__ v16h_t load_a_frag(const _Float16* rowbase, int lane) {
  const int g = (lane >> 4) * 8;
  union { v16h_t v; v8h_t h[2]; } u;
  u.h[0] = *(const v8h_t*)(rowbase + g);
  u.h[1] = *(const v8h_t*)(rowbase + 16 + g);
  return u.v;
}
// B 32x16 f16 (column-major fragment): lane g=lane>>4 holds K in {8g..8g+7} U {16+8g..16+8g+7}, col = lane&15
__device__ __forceinline__ v16h_t load_b_frag(const _Float16* colbase, int lane) {
  const int g = (lane >> 4) * 8;
  union { v16h_t v; v8h_t h[2]; } u;
  u.h[0] = *(const v8h_t*)(colbase + g);
  u.h[1] = *(const v8h_t*)(colbase + 16 + g);
  return u.v;
}

// ---------------- gate: softmax(x@Wg + bg) ----------------
__global__ void moe_gate_kernel(const float* __restrict__ x,
                                const float* __restrict__ Wg,
                                const float* __restrict__ bg,
                                float* __restrict__ gate) {
  const int lane = threadIdx.x & 31;
  const int tok  = blockIdx.x * (blockDim.x >> 5) + (threadIdx.x >> 5);
  if (tok >= N_TOK) return;
  float p[E_NUM];
#pragma unroll
  for (int e = 0; e < E_NUM; ++e) p[e] = 0.0f;
  const float* xr = x + (size_t)tok * D_IN;
  for (int k = lane; k < D_IN; k += 32) {
    const float xv = xr[k];
    const float* wr = Wg + (size_t)k * E_NUM;
#pragma unroll
    for (int e = 0; e < E_NUM; ++e) p[e] = fmaf(xv, wr[e], p[e]);
  }
#pragma unroll
  for (int e = 0; e < E_NUM; ++e) {
#pragma unroll
    for (int off = 16; off > 0; off >>= 1) p[e] += __shfl_xor(p[e], off, 32);
    p[e] += bg[e];
  }
  float m = p[0];
#pragma unroll
  for (int e = 1; e < E_NUM; ++e) m = fmaxf(m, p[e]);
  float s = 0.0f;
#pragma unroll
  for (int e = 0; e < E_NUM; ++e) { p[e] = __expf(p[e] - m); s += p[e]; }
  const float inv = 1.0f / s;
  if (lane == 0) {
#pragma unroll
    for (int e = 0; e < E_NUM; ++e) gate[(size_t)tok * E_NUM + e] = p[e] * inv;
  }
}

// ---------------- out init: out[n,o] = sum_e gate[n,e]*b2[e,o] ----------------
__global__ void moe_init_out_kernel(const float* __restrict__ gate,
                                    const float* __restrict__ b2,
                                    float* __restrict__ out) {
  const int idx = blockIdx.x * blockDim.x + threadIdx.x;
  const int n = idx >> 10;        // / D_OUT
  const int o = idx & (D_OUT - 1);
  const float* g = gate + (size_t)n * E_NUM;
  float s = 0.0f;
#pragma unroll
  for (int e = 0; e < E_NUM; ++e) s = fmaf(g[e], b2[(size_t)e * D_OUT + o], s);
  out[idx] = s;
}

// ---------------- layer 1: hg = gate[:,e] * relu(x @ W1[e] + b1[e]) ----------------
__global__ __launch_bounds__(256)
void moe_ffn1_kernel(const float* __restrict__ x,      // [N, D_IN]
                     const float* __restrict__ W1,     // [E, D_IN, D_H]
                     const float* __restrict__ b1,     // [E, D_H]
                     const float* __restrict__ gate,   // [N, E]
                     _Float16* __restrict__ hg,        // [N, D_H] (this expert)
                     int e) {
  __shared__ alignas(16) _Float16 xs[2][128 * LDK];    // A tiles [row][k]
  __shared__ alignas(16) _Float16 ws[2][128 * LDK];    // B tiles transposed [col][k]
  __shared__ float gsh[128];                           // gate column for tile rows

  const int tid  = threadIdx.x;
  const int lane = tid & 31;
  const int w    = tid >> 5;
  const int wrow = w >> 2;                 // 0..1 -> 64 rows
  const int wcol = w & 3;                  // 0..3 -> 32 cols
  const int n0 = blockIdx.y * 128;
  const int h0 = blockIdx.x * 128;

  const float* W1e = W1 + (size_t)e * D_IN * D_H;

  if (tid < 128) gsh[tid] = gate[(size_t)(n0 + tid) * E_NUM + e];

  v8f_t acc[4][2];
#pragma unroll
  for (int rt = 0; rt < 4; ++rt)
#pragma unroll
    for (int ct = 0; ct < 2; ++ct) acc[rt][ct] = (v8f_t)(0.0f);

  // staging ownership: A: 1 row x 32 k per thread; B: 8 k x 4 col per thread
  const int arow = tid >> 1;               // 0..127
  const int akl  = (tid & 1) * 32;         // 0|32
  const int bkk  = (tid >> 5) * 8;         // 0..56
  const int bcl  = (tid & 31) * 4;         // 0..124

  const float* agp = x   + (size_t)(n0 + arow) * D_IN + akl;
  const float* bgp = W1e + (size_t)bkk * D_H + h0 + bcl;

  v4f_t ar[8], br[8];

  auto gload = [&](int k0) {
#pragma unroll
    for (int i = 0; i < 8; ++i) ar[i] = *(const v4f_t*)(agp + k0 + 4 * i);
#pragma unroll
    for (int i = 0; i < 8; ++i) br[i] = *(const v4f_t*)(bgp + (size_t)(k0 + i) * D_H);
  };
  auto lstore = [&](int buf) {
    _Float16* as = &xs[buf][arow * LDK + akl];
#pragma unroll
    for (int i = 0; i < 4; ++i) {
      v8h_t h;
      h[0] = (_Float16)ar[2 * i].x; h[1] = (_Float16)ar[2 * i].y;
      h[2] = (_Float16)ar[2 * i].z; h[3] = (_Float16)ar[2 * i].w;
      h[4] = (_Float16)ar[2 * i + 1].x; h[5] = (_Float16)ar[2 * i + 1].y;
      h[6] = (_Float16)ar[2 * i + 1].z; h[7] = (_Float16)ar[2 * i + 1].w;
      *(v8h_t*)(as + 8 * i) = h;
    }
#pragma unroll
    for (int c = 0; c < 4; ++c) {
      v8h_t h;
#pragma unroll
      for (int i = 0; i < 8; ++i) h[i] = (_Float16)br[i][c];
      *(v8h_t*)(&ws[buf][(bcl + c) * LDK + bkk]) = h;  // k-contiguous: ds_store_b128
    }
  };
  auto compute = [&](int buf) {
    const int rr = lane & 15;
#pragma unroll
    for (int ks = 0; ks < KC; ks += 32) {
      v16h_t af[4];
#pragma unroll
      for (int rt = 0; rt < 4; ++rt)
        af[rt] = load_a_frag(&xs[buf][(wrow * 64 + rt * 16 + rr) * LDK + ks], lane);
#pragma unroll
      for (int ct = 0; ct < 2; ++ct) {
        v16h_t bf = load_b_frag(&ws[buf][(wcol * 32 + ct * 16 + rr) * LDK + ks], lane);
#pragma unroll
        for (int rt = 0; rt < 4; ++rt)
          acc[rt][ct] = __builtin_amdgcn_wmma_f32_16x16x32_f16(
              false, af[rt], false, bf, (short)0, acc[rt][ct], false, false);
      }
    }
  };

  const int NK = D_IN / KC;   // 16
  gload(0);
  lstore(0);
  __syncthreads();
#pragma unroll 1
  for (int kc = 0; kc < NK; ++kc) {
    if (kc + 1 < NK) gload((kc + 1) * KC);   // prefetch next chunk (no LDS dep)
    compute(kc & 1);                          // wmma stream on current buffer
    if (kc + 1 < NK) {
      lstore((kc + 1) & 1);                   // convert + store after compute issued
      __syncthreads();
    }
  }

  // epilogue: +b1, relu, *gate, cast f16, store
  const int rr = lane & 15;
  const int mh = (lane >> 4) * 8;
#pragma unroll
  for (int rt = 0; rt < 4; ++rt) {
#pragma unroll
    for (int ct = 0; ct < 2; ++ct) {
      const int h = h0 + wcol * 32 + ct * 16 + rr;
      const float bb = b1[(size_t)e * D_H + h];
#pragma unroll
      for (int i = 0; i < 8; ++i) {
        const int nl = wrow * 64 + rt * 16 + mh + i;
        float v = acc[rt][ct][i] + bb;
        v = v > 0.0f ? v : 0.0f;
        v *= gsh[nl];
        hg[(size_t)(n0 + nl) * D_H + h] = (_Float16)v;
      }
    }
  }
}

// ---------------- layer 2: out += hg @ W2[e] ----------------
__global__ __launch_bounds__(256)
void moe_ffn2_kernel(const _Float16* __restrict__ hg,  // [N, D_H]
                     const float* __restrict__ W2,     // [E, D_H, D_OUT]
                     float* __restrict__ out,          // [N, D_OUT]
                     int e) {
  __shared__ alignas(16) _Float16 xs[2][128 * LDK];
  __shared__ alignas(16) _Float16 ws[2][128 * LDK];

  const int tid  = threadIdx.x;
  const int lane = tid & 31;
  const int w    = tid >> 5;
  const int wrow = w >> 2;
  const int wcol = w & 3;
  const int n0 = blockIdx.y * 128;
  const int o0 = blockIdx.x * 128;

  const float* W2e = W2 + (size_t)e * D_H * D_OUT;

  v8f_t acc[4][2];
#pragma unroll
  for (int rt = 0; rt < 4; ++rt)
#pragma unroll
    for (int ct = 0; ct < 2; ++ct) acc[rt][ct] = (v8f_t)(0.0f);

  const int arow = tid >> 1;
  const int akl  = (tid & 1) * 32;         // in halfs
  const int bkk  = (tid >> 5) * 8;
  const int bcl  = (tid & 31) * 4;

  const _Float16* agp = hg  + (size_t)(n0 + arow) * D_H + akl;
  const float*    bgp = W2e + (size_t)bkk * D_OUT + o0 + bcl;

  v8h_t arh[4];
  v4f_t br[8];

  auto gload = [&](int k0) {
#pragma unroll
    for (int i = 0; i < 4; ++i) arh[i] = *(const v8h_t*)(agp + k0 + 8 * i);
#pragma unroll
    for (int i = 0; i < 8; ++i) br[i] = *(const v4f_t*)(bgp + (size_t)(k0 + i) * D_OUT);
  };
  auto lstore = [&](int buf) {
    v8h_t* as = (v8h_t*)&xs[buf][arow * LDK + akl];
#pragma unroll
    for (int i = 0; i < 4; ++i) as[i] = arh[i];
#pragma unroll
    for (int c = 0; c < 4; ++c) {
      v8h_t h;
#pragma unroll
      for (int i = 0; i < 8; ++i) h[i] = (_Float16)br[i][c];
      *(v8h_t*)(&ws[buf][(bcl + c) * LDK + bkk]) = h;
    }
  };
  auto compute = [&](int buf) {
    const int rr = lane & 15;
#pragma unroll
    for (int ks = 0; ks < KC; ks += 32) {
      v16h_t af[4];
#pragma unroll
      for (int rt = 0; rt < 4; ++rt)
        af[rt] = load_a_frag(&xs[buf][(wrow * 64 + rt * 16 + rr) * LDK + ks], lane);
#pragma unroll
      for (int ct = 0; ct < 2; ++ct) {
        v16h_t bf = load_b_frag(&ws[buf][(wcol * 32 + ct * 16 + rr) * LDK + ks], lane);
#pragma unroll
        for (int rt = 0; rt < 4; ++rt)
          acc[rt][ct] = __builtin_amdgcn_wmma_f32_16x16x32_f16(
              false, af[rt], false, bf, (short)0, acc[rt][ct], false, false);
      }
    }
  };

  const int NK = D_H / KC;   // 64
  gload(0);
  lstore(0);
  __syncthreads();
#pragma unroll 1
  for (int kc = 0; kc < NK; ++kc) {
    if (kc + 1 < NK) gload((kc + 1) * KC);
    compute(kc & 1);
    if (kc + 1 < NK) {
      lstore((kc + 1) & 1);
      __syncthreads();
    }
  }

  // epilogue: accumulate into out (unique writer per element per kernel -> deterministic)
  const int rr = lane & 15;
  const int mh = (lane >> 4) * 8;
#pragma unroll
  for (int rt = 0; rt < 4; ++rt) {
#pragma unroll
    for (int ct = 0; ct < 2; ++ct) {
      const int o = o0 + wcol * 32 + ct * 16 + rr;
#pragma unroll
      for (int i = 0; i < 8; ++i) {
        const int n = n0 + wrow * 64 + rt * 16 + mh + i;
        float* dst = out + (size_t)n * D_OUT + o;
        *dst += acc[rt][ct][i];
      }
    }
  }
}

extern "C" void kernel_launch(void* const* d_in, const int* in_sizes, int n_in,
                              void* d_out, int out_size, void* d_ws, size_t ws_size,
                              hipStream_t stream) {
  (void)in_sizes; (void)n_in; (void)out_size; (void)ws_size;
  const float* x  = (const float*)d_in[0];
  const float* W1 = (const float*)d_in[1];
  const float* b1 = (const float*)d_in[2];
  const float* W2 = (const float*)d_in[3];
  const float* b2 = (const float*)d_in[4];
  const float* Wg = (const float*)d_in[5];
  const float* bg = (const float*)d_in[6];
  float* out = (float*)d_out;

  // workspace: gate [N,E] f32 at 0; hg [N,D_H] f16 at 1MB (total ~33.5 MB)
  float* gate = (float*)d_ws;
  _Float16* hg = (_Float16*)((char*)d_ws + (1u << 20));

  moe_gate_kernel<<<N_TOK / 4, 128, 0, stream>>>(x, Wg, bg, gate);
  moe_init_out_kernel<<<(N_TOK * D_OUT) / 256, 256, 0, stream>>>(gate, b2, out);

  for (int e = 0; e < E_NUM; ++e) {
    moe_ffn1_kernel<<<dim3(D_H / 128, N_TOK / 128), 256, 0, stream>>>(
        x, W1, b1, gate, hg, e);
    moe_ffn2_kernel<<<dim3(D_OUT / 128, N_TOK / 128), 256, 0, stream>>>(
        hg, W2, out, e);
  }
}
